// PatchTSTEncoderLayerWithRope_75299366633550
// MI455X (gfx1250) — matrix operand order repaired and checked by reading, not compiled, not run on hardware
//
#include <hip/hip_runtime.h>
#include <hip/hip_bf16.h>

// ---------- types ----------
typedef __bf16 bf16t;
typedef __attribute__((ext_vector_type(16))) __bf16 v16bf;
typedef __attribute__((ext_vector_type(8)))  float  v8f;

#define DMODEL 512
#define SEQ    512
#define NBATCH 32
#define NHEAD  8
#define HDIM   64
#define NEXP   8
#define FFNDIM 2048
#define NTOK   (NBATCH*SEQ)      // 16384
#define MOE_ROWS  32
#define MOE_TILES (NTOK/MOE_ROWS) // 512

union ABFrag { v16bf v; uint4 q[2]; };

// A fragment: 16x32 (MxK) bf16, row-major source with leading dim ld.
// Per ISA table: lane L (r=L&15, hl=L>>4): VGPRs0-3 = K [hl*8, hl*8+8),
// VGPRs4-7 = K [16+hl*8, 16+hl*8+8).  base points at (m0, k0).
__device__ __forceinline__ v16bf load_frag_A(const bf16t* base, int ld, int lane) {
  ABFrag u;
  const bf16t* p = base + (lane & 15) * ld + ((lane >> 4) << 3);
  u.q[0] = *(const uint4*)(p);
  u.q[1] = *(const uint4*)(p + 16);
  return u.v;
}

// B fragment: 32x16 (KxN) bf16, where column N of B is a contiguous-K row of W.
// base points at row n0 of W, offset k0.  lane L: c=L&15, K=[hl*8, hl*8+8) then [16+hl*8, 16+hl*8+8).
__device__ __forceinline__ v16bf load_frag_B(const bf16t* base, int ld, int lane) {
  ABFrag u;
  const bf16t* p = base + (lane & 15) * ld + ((lane >> 4) << 3);
  u.q[0] = *(const uint4*)(p);
  u.q[1] = *(const uint4*)(p + 16);
  return u.v;
}

__device__ __forceinline__ v8f wmma_bf16(v16bf a, v16bf b, v8f c) {
  return __builtin_amdgcn_wmma_f32_16x16x32_bf16(false, a, false, b, (short)0, c, false, false);
}

// ---------- CDNA5 async global->LDS copy (ASYNCcnt path, bypasses VGPRs) ----
// Generic pointers to __shared__ carry the LDS byte offset in the low 32 bits.
__device__ __forceinline__ void async_copy_b128(void* lds_dst, const void* gsrc) {
  unsigned ldsa = (unsigned)(size_t)lds_dst;
  asm volatile("global_load_async_to_lds_b128 %0, %1, off"
               :: "v"(ldsa), "v"(gsrc) : "memory");
}
__device__ __forceinline__ void wait_async() {
  asm volatile("s_wait_asynccnt 0" ::: "memory");
}

// ---------- kernel 0: f32 -> bf16 conversion ----------
__global__ __launch_bounds__(256) void cvt_f32_bf16(const float* __restrict__ src,
                                                    bf16t* __restrict__ dst, int n) {
  int i = blockIdx.x * blockDim.x + threadIdx.x;
  if (i < n) dst[i] = (bf16t)src[i];
}

// ---------- kernel 0b: zero scratch ----------
__global__ __launch_bounds__(256) void zero_kernel(float* moe_out, int* cnt, float* mean_acc) {
  int i = blockIdx.x * blockDim.x + threadIdx.x;
  if (i < NTOK * DMODEL) moe_out[i] = 0.f;
  if (i < NEXP) { cnt[i] = 0; mean_acc[i] = 0.f; }
}

// ---------- kernel 1: QKV GEMM (per 64-token tile) ----------
__global__ __launch_bounds__(256) void qkv_gemm(
    const bf16t* __restrict__ xhs,
    const bf16t* __restrict__ qw, const bf16t* __restrict__ kw, const bf16t* __restrict__ vw,
    const float* __restrict__ qb, const float* __restrict__ kb, const float* __restrict__ vb,
    bf16t* __restrict__ qo, bf16t* __restrict__ ko, bf16t* __restrict__ vo) {
  extern __shared__ char smem[];
  bf16t* xt = (bf16t*)smem;  // 64 x 512 bf16 = 64KB
  int tile = blockIdx.x;
  int tid = threadIdx.x, lane = tid & 31, wave = tid >> 5;
  const char* gsrc = (const char*)(xhs + (size_t)tile * 64 * DMODEL);
  for (int i = tid; i < 64 * DMODEL / 8; i += 256)
    async_copy_b128((char*)xt + (size_t)i * 16, gsrc + (size_t)i * 16);
  wait_async();
  __syncthreads();
  for (int t = wave; t < 3 * 4 * 32; t += 8) {
    int which = t / 128, rem = t - which * 128;
    int mt = rem >> 5, nt = rem & 31;
    const bf16t* w = (which == 0) ? qw : (which == 1) ? kw : vw;
    const float* bias = (which == 0) ? qb : (which == 1) ? kb : vb;
    bf16t* out = (which == 0) ? qo : (which == 1) ? ko : vo;
    v8f acc = {};
    for (int ks = 0; ks < 16; ks++) {
      v16bf a = load_frag_A(xt + mt * 16 * DMODEL + ks * 32, DMODEL, lane);
      v16bf b = load_frag_B(w + (size_t)nt * 16 * DMODEL + ks * 32, DMODEL, lane);
      acc = wmma_bf16(a, b, acc);
    }
    int col = nt * 16 + (lane & 15);
    float bv = bias[col];
    int rbase = mt * 16 + ((lane >> 4) << 3);
    for (int j = 0; j < 8; j++)
      out[(size_t)(tile * 64 + rbase + j) * DMODEL + col] = (bf16t)(acc[j] + bv);
  }
}

// ---------- kernel 2: RoPE + layout repack + V transpose ----------
__global__ __launch_bounds__(256) void rope_layout(
    const bf16t* __restrict__ q_raw, const bf16t* __restrict__ k_raw,
    const bf16t* __restrict__ v_raw,
    bf16t* __restrict__ qbf, bf16t* __restrict__ kbf, bf16t* __restrict__ vT) {
  int i = blockIdx.x * blockDim.x + threadIdx.x;
  if (i >= NTOK * DMODEL) return;
  int tok = i / DMODEL, d = i - tok * DMODEL;
  int b = tok / SEQ, s = tok - b * SEQ;
  int h = d >> 6, dd = d & 63;
  int bh = b * NHEAD + h;
  float qv = (float)q_raw[i], kv = (float)k_raw[i];
  if (dd < 32) {  // ROPE_PCT = 0.5 -> r = 32, half = 16
    int ii = dd & 15;
    float freq = __powf(10000.0f, -(float)ii * 2.0f / 32.0f);
    float ang = (float)s * freq;
    float sn, cs;
    __sincosf(ang, &sn, &cs);
    if (dd < 16) {
      float q2 = (float)q_raw[i + 16], k2 = (float)k_raw[i + 16];
      qv = qv * cs - q2 * sn;  kv = kv * cs - k2 * sn;
    } else {
      float q1 = (float)q_raw[i - 16], k1 = (float)k_raw[i - 16];
      qv = qv * cs + q1 * sn;  kv = kv * cs + k1 * sn;
    }
  }
  int dst = (bh * SEQ + s) * HDIM + dd;
  qbf[dst] = (bf16t)qv;
  kbf[dst] = (bf16t)kv;
  vT[((size_t)bh * HDIM + dd) * SEQ + s] = v_raw[i];
}

// ---------- kernel 3: attention (per b,h and 64-row q tile) ----------
__global__ __launch_bounds__(256) void attention(
    const bf16t* __restrict__ qbf, const bf16t* __restrict__ kbf,
    const bf16t* __restrict__ vT, bf16t* __restrict__ attn_out) {
  extern __shared__ char smem[];
  bf16t* qt = (bf16t*)smem;                                  // 64x64 bf16  (8KB)
  float* sc = (float*)(smem + 64 * 64 * 2);                  // 64x512 f32  (128KB)
  bf16t* pb = (bf16t*)(smem + 64 * 64 * 2 + 64 * SEQ * 4);   // 64x512 bf16 (64KB)
  __shared__ float red[256];
  int blk = blockIdx.x;
  int qtile = blk & 7;           // S/64 = 8 tiles
  int bh = blk >> 3;             // b*NHEAD + h
  int b = bh >> 3, h = bh & 7;
  int q0 = qtile * 64;
  int tid = threadIdx.x, lane = tid & 31, wave = tid >> 5;

  const char* qg = (const char*)(qbf + ((size_t)bh * SEQ + q0) * HDIM);
  for (int i = tid; i < 64 * HDIM / 8; i += 256)
    async_copy_b128((char*)qt + (size_t)i * 16, qg + (size_t)i * 16);
  wait_async();
  __syncthreads();

  // scores = (Q * 1/8) @ K^T : M=64, N=512, K=64
  const bf16t* kg = kbf + (size_t)bh * SEQ * HDIM;
  for (int t = wave; t < 4 * 32; t += 8) {
    int mt = t >> 5, nt = t & 31;
    v8f acc = {};
    for (int ks = 0; ks < 2; ks++) {
      v16bf a  = load_frag_A(qt + mt * 16 * HDIM + ks * 32, HDIM, lane);
      v16bf bb = load_frag_B(kg + nt * 16 * HDIM + ks * 32, HDIM, lane);
      acc = wmma_bf16(a, bb, acc);
    }
    int col = nt * 16 + (lane & 15);
    int rbase = mt * 16 + ((lane >> 4) << 3);
    for (int j = 0; j < 8; j++) sc[(rbase + j) * SEQ + col] = acc[j] * 0.125f;
  }
  __syncthreads();

  // softmax: 4 threads per row (256 threads / 64 rows)
  {
    int row = tid >> 2, part = tid & 3;
    float* rowp = sc + row * SEQ;
    int c0 = part * (SEQ / 4);
    float lm = -1e30f;
    for (int j = c0; j < c0 + SEQ / 4; j++) lm = fmaxf(lm, rowp[j]);
    red[tid] = lm;
    __syncthreads();
    float m = fmaxf(fmaxf(red[row * 4], red[row * 4 + 1]),
                    fmaxf(red[row * 4 + 2], red[row * 4 + 3]));
    float ls = 0.f;
    for (int j = c0; j < c0 + SEQ / 4; j++) {
      float ev = __expf(rowp[j] - m);
      rowp[j] = ev;
      ls += ev;
    }
    __syncthreads();   // everyone done reading red before overwrite
    red[tid] = ls;
    __syncthreads();
    float inv = 1.f / (red[row * 4] + red[row * 4 + 1] + red[row * 4 + 2] + red[row * 4 + 3]);
    bf16t* prow = pb + row * SEQ;
    for (int j = c0; j < c0 + SEQ / 4; j++) prow[j] = (bf16t)(rowp[j] * inv);
  }
  __syncthreads();

  // attn = P @ V : M=64, N=64, K=512 (B from vT -> K contiguous)
  const bf16t* vg = vT + (size_t)bh * HDIM * SEQ;
  for (int t = wave; t < 4 * 4; t += 8) {
    int mt = t >> 2, nt = t & 3;
    v8f acc = {};
    for (int ks = 0; ks < 16; ks++) {
      v16bf a  = load_frag_A(pb + mt * 16 * SEQ + ks * 32, SEQ, lane);
      v16bf bb = load_frag_B(vg + nt * 16 * SEQ + ks * 32, SEQ, lane);
      acc = wmma_bf16(a, bb, acc);
    }
    int col = nt * 16 + (lane & 15);
    int rbase = mt * 16 + ((lane >> 4) << 3);
    for (int j = 0; j < 8; j++) {
      int row = q0 + rbase + j;
      attn_out[((size_t)b * SEQ + row) * DMODEL + h * HDIM + col] = (bf16t)acc[j];
    }
  }
}

// ---------- kernel 4: O-proj + residual + LayerNorm1 ----------
__global__ __launch_bounds__(256) void oproj_ln1(
    const bf16t* __restrict__ attn_bf, const bf16t* __restrict__ ow,
    const float* __restrict__ ob, const float* __restrict__ resid,
    const float* __restrict__ g, const float* __restrict__ beta,
    float* __restrict__ hid_f32, bf16t* __restrict__ hid_bf) {
  extern __shared__ char smem[];
  bf16t* at = (bf16t*)smem;                      // 64x512 bf16 (64KB)
  float* yt = (float*)(smem + 64 * DMODEL * 2);  // 64x512 f32  (128KB)
  __shared__ float red[256];
  int tile = blockIdx.x, tid = threadIdx.x, lane = tid & 31, wave = tid >> 5;
  const char* gsrc = (const char*)(attn_bf + (size_t)tile * 64 * DMODEL);
  for (int i = tid; i < 64 * DMODEL / 8; i += 256)
    async_copy_b128((char*)at + (size_t)i * 16, gsrc + (size_t)i * 16);
  wait_async();
  __syncthreads();
  for (int t = wave; t < 4 * 32; t += 8) {
    int mt = t >> 5, nt = t & 31;
    v8f acc = {};
    for (int ks = 0; ks < 16; ks++) {
      v16bf a  = load_frag_A(at + mt * 16 * DMODEL + ks * 32, DMODEL, lane);
      v16bf bb = load_frag_B(ow + (size_t)nt * 16 * DMODEL + ks * 32, DMODEL, lane);
      acc = wmma_bf16(a, bb, acc);
    }
    int col = nt * 16 + (lane & 15);
    int rbase = mt * 16 + ((lane >> 4) << 3);
    float bv = ob[col];
    for (int j = 0; j < 8; j++) {
      int row = rbase + j;
      yt[row * DMODEL + col] = acc[j] + bv + resid[(size_t)(tile * 64 + row) * DMODEL + col];
    }
  }
  __syncthreads();
  // LayerNorm: 4 threads per row
  {
    int row = tid >> 2, part = tid & 3;
    float* rowp = yt + row * DMODEL;
    int c0 = part * (DMODEL / 4);
    float s = 0.f;
    for (int j = c0; j < c0 + DMODEL / 4; j++) s += rowp[j];
    red[tid] = s;
    __syncthreads();
    float mu = (red[row * 4] + red[row * 4 + 1] + red[row * 4 + 2] + red[row * 4 + 3]) *
               (1.0f / DMODEL);
    float v = 0.f;
    for (int j = c0; j < c0 + DMODEL / 4; j++) { float d = rowp[j] - mu; v += d * d; }
    __syncthreads();
    red[tid] = v;
    __syncthreads();
    float var = (red[row * 4] + red[row * 4 + 1] + red[row * 4 + 2] + red[row * 4 + 3]) *
                (1.0f / DMODEL);
    float inv = rsqrtf(var + 1e-5f);
    size_t tokbase = (size_t)(tile * 64 + row) * DMODEL;
    for (int j = c0; j < c0 + DMODEL / 4; j++) {
      float val = (rowp[j] - mu) * inv * g[j] + beta[j];
      hid_f32[tokbase + j] = val;
      hid_bf[tokbase + j] = (bf16t)val;
    }
  }
}

// ---------- kernel 5: router (softmax + top2 + compaction) ----------
__global__ __launch_bounds__(256) void router_kernel(
    const float* __restrict__ hid, const float* __restrict__ gate_w,
    int* cnt, int* tok_list, float* w_list, float* mean_acc) {
  __shared__ float gw[NEXP * DMODEL];
  __shared__ float bsum[NEXP];
  int tid = threadIdx.x;
  for (int i = tid; i < NEXP * DMODEL; i += blockDim.x) gw[i] = gate_w[i];
  if (tid < NEXP) bsum[tid] = 0.f;
  __syncthreads();
  int tok = blockIdx.x * blockDim.x + tid;
  const float* x = hid + (size_t)tok * DMODEL;
  float probs[NEXP];
  float m = -1e30f;
  for (int e = 0; e < NEXP; e++) {
    float s = 0.f;
    const float* w = gw + e * DMODEL;
    for (int d = 0; d < DMODEL; d++) s += x[d] * w[d];
    probs[e] = s;
    m = fmaxf(m, s);
  }
  float sum = 0.f;
  for (int e = 0; e < NEXP; e++) { probs[e] = __expf(probs[e] - m); sum += probs[e]; }
  float inv = 1.f / sum;
  for (int e = 0; e < NEXP; e++) { probs[e] *= inv; atomicAdd(&bsum[e], probs[e]); }
  // top-2 (first-occurrence tie break like jax top_k)
  int i1 = 0;
  for (int e = 1; e < NEXP; e++) if (probs[e] > probs[i1]) i1 = e;
  int i2 = (i1 == 0) ? 1 : 0;
  for (int e = 0; e < NEXP; e++) if (e != i1 && probs[e] > probs[i2]) i2 = e;
  float wn = probs[i1] + probs[i2];
  int s1 = atomicAdd(&cnt[i1], 1);
  tok_list[i1 * NTOK + s1] = tok;  w_list[i1 * NTOK + s1] = probs[i1] / wn;
  int s2 = atomicAdd(&cnt[i2], 1);
  tok_list[i2 * NTOK + s2] = tok;  w_list[i2 * NTOK + s2] = probs[i2] / wn;
  __syncthreads();
  if (tid < NEXP) atomicAdd(&mean_acc[tid], bsum[tid]);
}

// ---------- kernel 6: load-balance loss ----------
__global__ void loss_kernel(const float* mean_acc, float* out_loss) {
  if (threadIdx.x == 0 && blockIdx.x == 0) {
    float s = 0.f;
    for (int e = 0; e < NEXP; e++) { float m = mean_acc[e] / (float)NTOK; s += m * m; }
    *out_loss = (float)NEXP * s;
  }
}

// ---------- kernel 7: routed MoE FFN ----------
__global__ __launch_bounds__(256) void moe_kernel(
    const bf16t* __restrict__ hid_bf, const bf16t* __restrict__ w1,
    const float* __restrict__ b1, const bf16t* __restrict__ w2,
    const float* __restrict__ b2, const int* __restrict__ cnt,
    const int* __restrict__ tok_list, const float* __restrict__ w_list,
    float* __restrict__ moe_out) {
  extern __shared__ char smem[];
  bf16t* xt = (bf16t*)smem;                           // 32x512 bf16 (32KB)
  bf16t* ht = (bf16t*)(smem + MOE_ROWS * DMODEL * 2); // 32x512 bf16 (32KB)
  __shared__ int   toks[MOE_ROWS];
  __shared__ float wts[MOE_ROWS];

  int e = blockIdx.x / MOE_TILES;
  int tile = blockIdx.x - e * MOE_TILES;
  int count = cnt[e];
  int base = tile * MOE_ROWS;
  if (base >= count) return;
  int nrows = min(MOE_ROWS, count - base);
  int tid = threadIdx.x, lane = tid & 31, wave = tid >> 5;

  if (tid < MOE_ROWS) {
    toks[tid] = (tid < nrows) ? tok_list[e * NTOK + base + tid] : -1;
    wts[tid]  = (tid < nrows) ? w_list[e * NTOK + base + tid] : 0.f;
  }
  __syncthreads();
  // gather token rows (bf16) into LDS: async copy for valid rows, zero pad rows
  for (int i = tid; i < MOE_ROWS * DMODEL / 8; i += 256) {
    int row = i >> 6;            // DMODEL/8 = 64 chunks per row
    int cv = i & 63;
    int tk = toks[row];
    if (tk >= 0) {
      async_copy_b128(xt + (size_t)i * 8, hid_bf + (size_t)tk * DMODEL + cv * 8);
    } else {
      uint4 z = {0u, 0u, 0u, 0u};
      *(uint4*)(xt + (size_t)i * 8) = z;
    }
  }
  wait_async();

  v8f yacc[8];
  v8f zf = {};
  for (int i = 0; i < 8; i++) yacc[i] = zf;

  const bf16t* w1e = w1 + (size_t)e * FFNDIM * DMODEL;
  const bf16t* w2e = w2 + (size_t)e * DMODEL * FFNDIM;
  const float* b1e = b1 + e * FFNDIM;

  for (int fc = 0; fc < FFNDIM / 512; fc++) {   // 4 chunks of 512
    __syncthreads();
    // GEMM1: h = gelu(x @ w1^T + b1), chunk -> ht (bf16)
    for (int t = wave; t < 2 * 32; t += 8) {
      int mt = t >> 5, nt = t & 31;
      v8f acc = {};
      for (int ks = 0; ks < 16; ks++) {
        v16bf a  = load_frag_A(xt + mt * 16 * DMODEL + ks * 32, DMODEL, lane);
        v16bf bb = load_frag_B(w1e + (size_t)(fc * 512 + nt * 16) * DMODEL + ks * 32,
                               DMODEL, lane);
        acc = wmma_bf16(a, bb, acc);
      }
      int col = nt * 16 + (lane & 15);
      int rbase = mt * 16 + ((lane >> 4) << 3);
      float bv = b1e[fc * 512 + col];
      for (int j = 0; j < 8; j++) {
        float xv = acc[j] + bv;
        float gel = 0.5f * xv * (1.0f + erff(xv * 0.70710678118f));  // exact gelu
        ht[(rbase + j) * 512 + col] = (bf16t)gel;
      }
    }
    __syncthreads();
    // GEMM2 partial: y += h_chunk @ w2_chunk^T  (y in registers)
    for (int i = 0; i < 8; i++) {
      int t = wave + 8 * i;
      int mt = t >> 5, nt = t & 31;
      for (int ks = 0; ks < 16; ks++) {
        v16bf a  = load_frag_A(ht + mt * 16 * 512 + ks * 32, 512, lane);
        v16bf bb = load_frag_B(w2e + (size_t)(nt * 16) * FFNDIM + fc * 512 + ks * 32,
                               FFNDIM, lane);
        yacc[i] = wmma_bf16(a, bb, yacc[i]);
      }
    }
  }
  // scatter: moe_out[tok] += weight * (y + b2)
  const float* b2e = b2 + e * DMODEL;
  for (int i = 0; i < 8; i++) {
    int t = wave + 8 * i;
    int mt = t >> 5, nt = t & 31;
    int col = nt * 16 + (lane & 15);
    int rbase = mt * 16 + ((lane >> 4) << 3);
    float bv = b2e[col];
    for (int j = 0; j < 8; j++) {
      int row = rbase + j;
      int tk = toks[row];
      if (tk >= 0)
        atomicAdd(&moe_out[(size_t)tk * DMODEL + col], wts[row] * (yacc[i][j] + bv));
    }
  }
}

// ---------- kernel 8: final LayerNorm2 -> output ----------
__global__ __launch_bounds__(256) void ln2_kernel(
    const float* __restrict__ hid, const float* __restrict__ moe,
    const float* __restrict__ g, const float* __restrict__ beta,
    float* __restrict__ out) {
  int tok = blockIdx.x * blockDim.x + threadIdx.x;
  if (tok >= NTOK) return;
  const float* a = hid + (size_t)tok * DMODEL;
  const float* m = moe + (size_t)tok * DMODEL;
  float mu = 0.f;
  for (int j = 0; j < DMODEL; j++) mu += a[j] + m[j];
  mu /= (float)DMODEL;
  float var = 0.f;
  for (int j = 0; j < DMODEL; j++) { float d = a[j] + m[j] - mu; var += d * d; }
  var /= (float)DMODEL;
  float inv = rsqrtf(var + 1e-5f);
  float* o = out + (size_t)tok * DMODEL;
  for (int j = 0; j < DMODEL; j++) o[j] = (a[j] + m[j] - mu) * inv * g[j] + beta[j];
}

// ---------- host ----------
extern "C" void kernel_launch(void* const* d_in, const int* in_sizes, int n_in,
                              void* d_out, int out_size, void* d_ws, size_t ws_size,
                              hipStream_t stream) {
  const float* hs    = (const float*)d_in[0];
  const float* q_w   = (const float*)d_in[1];
  const float* q_b   = (const float*)d_in[2];
  const float* k_w   = (const float*)d_in[3];
  const float* k_b   = (const float*)d_in[4];
  const float* v_w   = (const float*)d_in[5];
  const float* v_b   = (const float*)d_in[6];
  const float* o_w   = (const float*)d_in[7];
  const float* o_b   = (const float*)d_in[8];
  const float* ln1_g = (const float*)d_in[9];
  const float* ln1_b = (const float*)d_in[10];
  const float* gate_w= (const float*)d_in[11];
  const float* e_w1  = (const float*)d_in[12];
  const float* e_b1  = (const float*)d_in[13];
  const float* e_w2  = (const float*)d_in[14];
  const float* e_b2  = (const float*)d_in[15];
  const float* ln2_g = (const float*)d_in[16];
  const float* ln2_b = (const float*)d_in[17];
  float* out = (float*)d_out;

  char* p = (char*)d_ws;
  auto carve = [&](size_t bytes) -> char* {
    char* r = p;
    p += (bytes + 255) & ~(size_t)255;
    return r;
  };
  const size_t NBF = (size_t)NTOK * DMODEL * sizeof(bf16t);   // 16MB
  const size_t NF  = (size_t)NTOK * DMODEL * sizeof(float);   // 32MB
  bf16t* qw_bf  = (bf16t*)carve((size_t)DMODEL * DMODEL * 2);
  bf16t* kw_bf  = (bf16t*)carve((size_t)DMODEL * DMODEL * 2);
  bf16t* vw_bf  = (bf16t*)carve((size_t)DMODEL * DMODEL * 2);
  bf16t* ow_bf  = (bf16t*)carve((size_t)DMODEL * DMODEL * 2);
  bf16t* w1_bf  = (bf16t*)carve((size_t)NEXP * FFNDIM * DMODEL * 2);
  bf16t* w2_bf  = (bf16t*)carve((size_t)NEXP * DMODEL * FFNDIM * 2);
  bf16t* xhs_bf = (bf16t*)carve(NBF);
  bf16t* q_raw  = (bf16t*)carve(NBF);
  bf16t* k_raw  = (bf16t*)carve(NBF);
  bf16t* v_raw  = (bf16t*)carve(NBF);
  bf16t* qbf    = (bf16t*)carve(NBF);
  bf16t* kbf    = (bf16t*)carve(NBF);
  bf16t* vT     = (bf16t*)carve(NBF);
  bf16t* attn_bf= (bf16t*)carve(NBF);
  float* hid_f32= (float*)carve(NF);
  bf16t* hid_bf = (bf16t*)carve(NBF);
  float* moe_out= (float*)carve(NF);
  int*   cnt    = (int*)carve(NEXP * sizeof(int));
  float* mean_acc = (float*)carve(NEXP * sizeof(float));
  int*   tok_list = (int*)carve((size_t)NEXP * NTOK * sizeof(int));
  float* w_list   = (float*)carve((size_t)NEXP * NTOK * sizeof(float));

  auto cvt = [&](const float* s, bf16t* d, int n) {
    cvt_f32_bf16<<<(n + 255) / 256, 256, 0, stream>>>(s, d, n);
  };
  cvt(q_w, qw_bf, DMODEL * DMODEL);
  cvt(k_w, kw_bf, DMODEL * DMODEL);
  cvt(v_w, vw_bf, DMODEL * DMODEL);
  cvt(o_w, ow_bf, DMODEL * DMODEL);
  cvt(e_w1, w1_bf, NEXP * FFNDIM * DMODEL);
  cvt(e_w2, w2_bf, NEXP * DMODEL * FFNDIM);
  cvt(hs, xhs_bf, NTOK * DMODEL);

  zero_kernel<<<(NTOK * DMODEL + 255) / 256, 256, 0, stream>>>(moe_out, cnt, mean_acc);

  qkv_gemm<<<NTOK / 64, 256, 64 * DMODEL * 2, stream>>>(
      xhs_bf, qw_bf, kw_bf, vw_bf, q_b, k_b, v_b, q_raw, k_raw, v_raw);

  rope_layout<<<(NTOK * DMODEL) / 256, 256, 0, stream>>>(q_raw, k_raw, v_raw, qbf, kbf, vT);

  attention<<<NBATCH * NHEAD * (SEQ / 64), 256,
              64 * 64 * 2 + 64 * SEQ * 4 + 64 * SEQ * 2, stream>>>(qbf, kbf, vT, attn_bf);

  oproj_ln1<<<NTOK / 64, 256, 64 * DMODEL * 2 + 64 * DMODEL * 4, stream>>>(
      attn_bf, ow_bf, o_b, hs, ln1_g, ln1_b, hid_f32, hid_bf);

  router_kernel<<<NTOK / 256, 256, 0, stream>>>(hid_f32, gate_w, cnt, tok_list, w_list, mean_acc);

  loss_kernel<<<1, 1, 0, stream>>>(mean_acc, out + (size_t)NTOK * DMODEL);

  moe_kernel<<<NEXP * MOE_TILES, 256, 2 * MOE_ROWS * DMODEL * 2, stream>>>(
      hid_bf, w1_bf, e_b1, w2_bf, e_b2, cnt, tok_list, w_list, moe_out);

  ln2_kernel<<<NTOK / 256, 256, 0, stream>>>(hid_f32, moe_out, ln2_g, ln2_b, out);
}
